// DMF_81552839017131
// MI455X (gfx1250) — matrix operand order repaired and checked by reading, not compiled, not run on hardware
//
#include <hip/hip_runtime.h>
#include <stdint.h>

// ---------------------------------------------------------------------------
// CDNA5 (gfx1250) bf16 WMMA helpers
// ---------------------------------------------------------------------------
typedef __attribute__((ext_vector_type(16))) __bf16 v16bf;
typedef __attribute__((ext_vector_type(8)))  float  v8f;
typedef __attribute__((ext_vector_type(4)))  unsigned int u32x4;
typedef __attribute__((ext_vector_type(4)))  int i32x4;
typedef unsigned short u16;

union BF16x16 { v16bf v; u16 u[16]; unsigned int d[8]; };

#if defined(__gfx1250__) && __has_builtin(__builtin_amdgcn_global_load_async_to_lds_b128) && __has_builtin(__builtin_amdgcn_s_wait_asynccnt)
#define HAVE_ASYNC_LDS 1
// builtin signature (from compiler diagnostics): (v4i32 as1*, v4i32 as3*, imm offset, imm cpol)
typedef __attribute__((address_space(1))) i32x4* g128_t;
typedef __attribute__((address_space(3))) i32x4* l128_t;
#else
#define HAVE_ASYNC_LDS 0
#endif

__device__ __forceinline__ u16 f2bf_u(float f) {
    union { float f; unsigned u; } x; x.f = f;
    unsigned r = (x.u + 0x7FFFu + ((x.u >> 16) & 1u)) >> 16;  // RNE
    return (u16)r;
}

// Convert two f32 -> packed bf16 pair (low = a, high = b): one v_perm_b32.
__device__ __forceinline__ unsigned cvt2(float a, float b) {
    union { float f; unsigned u; } xa, xb; xa.f = a; xb.f = b;
    return __builtin_amdgcn_perm(xb.u, xa.u, 0x07060302u);
}

__device__ __forceinline__ float bflo_f(unsigned u) {
    union { unsigned x; float f; } z; z.x = u << 16; return z.f;
}
__device__ __forceinline__ float bfhi_f(unsigned u) {
    union { unsigned x; float f; } z; z.x = u & 0xFFFF0000u; return z.f;
}
__device__ __forceinline__ float bf2f(u16 h) {
    union { unsigned x; float f; } z; z.x = ((unsigned)h) << 16; return z.f;
}

__device__ __forceinline__ v8f wmma_bf16(v16bf a, v16bf b, v8f c) {
    return __builtin_amdgcn_wmma_f32_16x16x32_bf16(
        false, a, false, b, (short)0, c, false, false);
}

// Load one 16-element bf16 fragment from two contiguous 16B chunks.
__device__ __forceinline__ v16bf ld_frag2(const u16* p0, const u16* p1) {
    BF16x16 f;
    *(u32x4*)&f.d[0] = *(const u32x4*)p0;
    *(u32x4*)&f.d[4] = *(const u32x4*)p1;
    return f.v;
}

// ---------------------------------------------------------------------------
// GEMM: C[b] (MxN) = W (MxK,f32) * X[b] (KxN), bf16 WMMA core, f32 accum.
// IN_BF16: X is bf16 (no conversion needed); OUT_BF16: C stored as bf16.
// grid: (N/128, M/128, B), block 256 (8 waves), each wave = 2x4 WMMA tiles.
// B tile stored TRANSPOSED in LDS (paired u32 writes); fragments = b128 loads.
// K-loop software-pipelined: next tile prefetched to registers during WMMA.
// ---------------------------------------------------------------------------
template <int IN_BF16, int OUT_BF16>
__global__ void __launch_bounds__(256)
gemm_w_x(const float* __restrict__ W, const void* __restrict__ Xv,
         void* __restrict__ Cv, int M, int K, int N) {
    __shared__ __align__(16) u16 As[128][40];    // [m][k]
    __shared__ __align__(16) u16 Bs_t[128][40];  // [n][k] (transposed)

    const float* Xf = (const float*)Xv;
    const u16*   Xh = (const u16*)Xv;
    Xf += (size_t)blockIdx.z * K * N;
    Xh += (size_t)blockIdx.z * K * N;

    const int tid  = threadIdx.x;
    const int lane = tid & 31;
    const int wave = tid >> 5;
    const int n0 = blockIdx.x * 128;
    const int m0 = blockIdx.y * 128;

    const int wy = wave >> 1;   // 0..3 : m-offset wy*32
    const int wx = wave & 1;    // 0..1 : n-offset wx*64
    const int l16 = lane & 15;
    const int lh  = lane >> 4;

    const int ar = tid >> 1,  akh = (tid & 1) * 16;  // A map: row, k-half
    const int bk = (tid >> 4) * 2;                   // B map: k-row pair
    const int bn = (tid & 15) * 8;                   // B map: n offset

    v8f zero = {};
    v8f acc[2][4];
    #pragma unroll
    for (int i = 0; i < 2; ++i)
        #pragma unroll
        for (int j = 0; j < 4; ++j) acc[i][j] = zero;

    float aR[16];
    float b0R[8], b1R[8];     // f32 input path
    u16   h0R[8], h1R[8];     // bf16 input path

    auto prefetch = [&](int kk) {
        const float* sa = W + (size_t)(m0 + ar) * K + kk + akh;
        #pragma unroll
        for (int i = 0; i < 16; ++i) aR[i] = sa[i];
        if constexpr (IN_BF16) {
            const u16* s0 = Xh + (size_t)(kk + bk) * N + n0 + bn;
            const u16* s1 = s0 + N;
            #pragma unroll
            for (int i = 0; i < 8; ++i) { h0R[i] = s0[i]; h1R[i] = s1[i]; }
        } else {
            const float* s0 = Xf + (size_t)(kk + bk) * N + n0 + bn;
            const float* s1 = s0 + N;
            #pragma unroll
            for (int i = 0; i < 8; ++i) { b0R[i] = s0[i]; b1R[i] = s1[i]; }
        }
    };
    prefetch(0);

    for (int k0 = 0; k0 < K; k0 += 32) {
        // commit prefetched tile to LDS as packed bf16 pairs
        #pragma unroll
        for (int j = 0; j < 8; ++j)
            *(unsigned*)&As[ar][akh + 2 * j] = cvt2(aR[2 * j], aR[2 * j + 1]);
        if constexpr (IN_BF16) {
            #pragma unroll
            for (int j = 0; j < 8; ++j)
                *(unsigned*)&Bs_t[bn + j][bk] =
                    (unsigned)h0R[j] | ((unsigned)h1R[j] << 16);
        } else {
            #pragma unroll
            for (int j = 0; j < 8; ++j)
                *(unsigned*)&Bs_t[bn + j][bk] = cvt2(b0R[j], b1R[j]);
        }
        __syncthreads();

        if (k0 + 32 < K) prefetch(k0 + 32);   // overlap with WMMAs

        BF16x16 afrag[2], bfrag[4];
        #pragma unroll
        for (int i2 = 0; i2 < 2; ++i2) {
            int row = wy * 32 + i2 * 16 + l16;
            afrag[i2].v = ld_frag2(&As[row][lh * 8], &As[row][16 + lh * 8]);
        }
        #pragma unroll
        for (int j = 0; j < 4; ++j) {
            int col = wx * 64 + j * 16 + l16;
            bfrag[j].v = ld_frag2(&Bs_t[col][lh * 8], &Bs_t[col][16 + lh * 8]);
        }
        #pragma unroll
        for (int i2 = 0; i2 < 2; ++i2)
            #pragma unroll
            for (int j = 0; j < 4; ++j)
                acc[i2][j] = wmma_bf16(afrag[i2].v, bfrag[j].v, acc[i2][j]);
        __syncthreads();
    }

    #pragma unroll
    for (int i2 = 0; i2 < 2; ++i2) {
        #pragma unroll
        for (int j = 0; j < 4; ++j) {
            int n  = n0 + wx * 64 + j * 16 + l16;
            int mb = m0 + wy * 32 + i2 * 16 + lh * 8;
            if constexpr (OUT_BF16) {
                u16* C = (u16*)Cv + (size_t)blockIdx.z * M * N;
                #pragma unroll
                for (int r = 0; r < 8; ++r)
                    C[(size_t)(mb + r) * N + n] = f2bf_u(acc[i2][j][r]);
            } else {
                float* C = (float*)Cv + (size_t)blockIdx.z * M * N;
                #pragma unroll
                for (int r = 0; r < 8; ++r)
                    C[(size_t)(mb + r) * N + n] = acc[i2][j][r];
            }
        }
    }
}

// ---------------------------------------------------------------------------
// Kernel 2: depthwise 3x3, pad 1, 1152 channels. bf16 in / bf16 out.
// ---------------------------------------------------------------------------
__global__ void __launch_bounds__(256)
dwconv3x3(const u16* __restrict__ in, const float* __restrict__ w,
          u16* __restrict__ out) {
    const size_t idx = (size_t)blockIdx.x * 256 + threadIdx.x;
    const int x  = (int)(idx & 127);
    const int y  = (int)((idx >> 7) & 127);
    const int ch = (int)((idx >> 14) % 1152);
    const size_t plane = idx >> 14;
    const u16* ip = in + (plane << 14);
    const float* wp = w + ch * 9;

    float s = 0.f;
    #pragma unroll
    for (int dy = 0; dy < 3; ++dy) {
        int yy = y + dy - 1;
        if (yy < 0 || yy > 127) continue;
        #pragma unroll
        for (int dx = 0; dx < 3; ++dx) {
            int xx = x + dx - 1;
            if (xx < 0 || xx > 127) continue;
            s += bf2f(ip[yy * 128 + xx]) * wp[dy * 3 + dx];
        }
    }
    out[idx] = f2bf_u(s);
}

// ---------------------------------------------------------------------------
// Kernel 3a: per (b,head): S = q k^T via WMMA over K=16384 (8 waves split K).
// q/k are bf16 -> fragments are DIRECT 16B global loads (zero conversion);
// squared norms unpacked from the same registers. Top-7 masked softmax ->
// attn (48x64 bf16, zero-padded K). grid: 64 blocks, 256 threads.
// ---------------------------------------------------------------------------
__global__ void __launch_bounds__(256)
attn_score(const u16* __restrict__ qkv, const float* __restrict__ temperature,
           u16* __restrict__ attn_out) {
    const int head = blockIdx.x & 7;
    const int bi   = blockIdx.x >> 3;
    const int tid  = threadIdx.x;
    const int lane = tid & 31;
    const int wave = tid >> 5;
    const int l16  = lane & 15;
    const int lh   = lane >> 4;

    const u16* qp = qkv + ((size_t)bi * 1152 +       head * 48) * 16384;
    const u16* kp = qkv + ((size_t)bi * 1152 + 384 + head * 48) * 16384;

    __shared__ float S[48][48];
    __shared__ float qn[48], kn[48];
    __shared__ u16 attn_s[48][64];

    for (int i = tid; i < 48 * 48; i += 256) (&S[0][0])[i] = 0.f;
    for (int i = tid; i < 48 * 64; i += 256) (&attn_s[0][0])[i] = 0;
    if (tid < 48) { qn[tid] = 0.f; kn[tid] = 0.f; }
    __syncthreads();

    v8f zero = {};
    v8f acc[3][3];
    #pragma unroll
    for (int i = 0; i < 3; ++i)
        #pragma unroll
        for (int j = 0; j < 3; ++j) acc[i][j] = zero;

    float qss[3] = {0.f, 0.f, 0.f};
    float kss[3] = {0.f, 0.f, 0.f};

    const int kend = (wave + 1) * 2048;
    for (int ks = wave * 2048; ks < kend; ks += 32) {
        BF16x16 af[3], bf[3];
        #pragma unroll
        for (int mt = 0; mt < 3; ++mt) {
            const u16* row = qp + (size_t)(mt * 16 + l16) * 16384 + ks;
            *(u32x4*)&af[mt].d[0] = *(const u32x4*)(row + lh * 8);
            *(u32x4*)&af[mt].d[4] = *(const u32x4*)(row + 16 + lh * 8);
            #pragma unroll
            for (int j = 0; j < 8; ++j) {
                float lo = bflo_f(af[mt].d[j]), hi = bfhi_f(af[mt].d[j]);
                qss[mt] += lo * lo + hi * hi;
            }
        }
        #pragma unroll
        for (int nt = 0; nt < 3; ++nt) {
            const u16* col = kp + (size_t)(nt * 16 + l16) * 16384 + ks + lh * 8;
            *(u32x4*)&bf[nt].d[0] = *(const u32x4*)col;
            *(u32x4*)&bf[nt].d[4] = *(const u32x4*)(col + 16);
            #pragma unroll
            for (int j = 0; j < 8; ++j) {
                float lo = bflo_f(bf[nt].d[j]), hi = bfhi_f(bf[nt].d[j]);
                kss[nt] += lo * lo + hi * hi;
            }
        }
        #pragma unroll
        for (int mt = 0; mt < 3; ++mt)
            #pragma unroll
            for (int nt = 0; nt < 3; ++nt)
                acc[mt][nt] = wmma_bf16(af[mt].v, bf[nt].v, acc[mt][nt]);
    }

    // reduce partial Grams + fused squared-norms across waves
    #pragma unroll
    for (int mt = 0; mt < 3; ++mt)
        #pragma unroll
        for (int nt = 0; nt < 3; ++nt)
            #pragma unroll
            for (int r = 0; r < 8; ++r)
                atomicAdd(&S[mt * 16 + lh * 8 + r][nt * 16 + l16], acc[mt][nt][r]);
    #pragma unroll
    for (int mt = 0; mt < 3; ++mt) atomicAdd(&qn[mt * 16 + l16], qss[mt]);
    #pragma unroll
    for (int nt = 0; nt < 3; ++nt) atomicAdd(&kn[nt * 16 + l16], kss[nt]);
    __syncthreads();

    // per-row normalize * temperature, top-7, masked softmax
    if (tid < 48) {
        const float t  = temperature[head];
        const float iq = 1.f / fmaxf(sqrtf(qn[tid]), 1e-12f);
        float vals[48];
        for (int d = 0; d < 48; ++d)
            vals[d] = S[tid][d] * iq * (1.f / fmaxf(sqrtf(kn[d]), 1e-12f)) * t;

        unsigned long long used = 0ull;
        float sel[7]; int selid[7];
        for (int it = 0; it < 7; ++it) {
            float best = -3.4e38f; int bd = 0;
            for (int d = 0; d < 48; ++d)
                if (!((used >> d) & 1ull) && vals[d] > best) { best = vals[d]; bd = d; }
            used |= 1ull << bd; sel[it] = best; selid[it] = bd;
        }
        const float m = sel[0];
        float e[7], sum = 0.f;
        for (int it = 0; it < 7; ++it) { e[it] = __expf(sel[it] - m); sum += e[it]; }
        const float inv = 1.f / sum;
        for (int it = 0; it < 7; ++it) attn_s[tid][selid[it]] = f2bf_u(e[it] * inv);
    }
    __syncthreads();

    u16* dst = attn_out + (size_t)blockIdx.x * 48 * 64;
    for (int i = tid; i < 48 * 64; i += 256) dst[i] = (&attn_s[0][0])[i];
}

// ---------------------------------------------------------------------------
// Kernel 3b: O = attn (48x48) @ v (48x16384), per (n-tile, head, b).
// grid: (128, 8, 8), block 256. K padded to 64. v is bf16; V tile transposed
// with paired u32 writes. attn staged via CDNA5 async global->LDS copy.
// Output stored bf16 for the proj GEMM.
// ---------------------------------------------------------------------------
__global__ void __launch_bounds__(256)
attn_apply(const u16* __restrict__ qkv, const u16* __restrict__ attn_buf,
           u16* __restrict__ O) {
    const int n0   = blockIdx.x * 128;
    const int head = blockIdx.y;
    const int bi   = blockIdx.z;
    const int tid  = threadIdx.x;
    const int lane = tid & 31;
    const int wave = tid >> 5;
    const int l16  = lane & 15;
    const int lh   = lane >> 4;

    const u16* vp = qkv + ((size_t)bi * 1152 + 768 + head * 48) * 16384;
    const u16* ap = attn_buf + (size_t)(bi * 8 + head) * 48 * 64;

    __shared__ __align__(16) u16 Asm[48][64];   // [c][k]
    __shared__ __align__(16) u16 Vs_t[128][72]; // [n][k] transposed

#if HAVE_ASYNC_LDS
    // raw 6 KB copy: 192 threads x 2 x 16B async global->LDS transfers
    if (tid < 192) {
        const u16* g = ap + tid * 16;                 // 32 B / thread
        u16* l = &Asm[0][0] + tid * 16;
        __builtin_amdgcn_global_load_async_to_lds_b128((g128_t)g, (l128_t)l, 0, 0);
        __builtin_amdgcn_global_load_async_to_lds_b128((g128_t)g, (l128_t)l, 16, 0);
    }
#else
    for (int i = tid; i < 48 * 64; i += 256) (&Asm[0][0])[i] = ap[i];
#endif

    // V tile: thread owns one n column, 24 consecutive d values -> 12 u32 pairs
    {
        const int vn  = tid & 127;
        const int vd0 = (tid >> 7) * 24;
        #pragma unroll
        for (int j = 0; j < 12; ++j) {
            int d = vd0 + 2 * j;
            unsigned lo = vp[(size_t)d * 16384 + n0 + vn];
            unsigned hi = vp[(size_t)(d + 1) * 16384 + n0 + vn];
            *(unsigned*)&Vs_t[vn][d] = lo | (hi << 16);
        }
    }
    for (int i = tid; i < 128 * 8; i += 256) {
        int n = i >> 3, k2 = 48 + 2 * (i & 7);
        *(unsigned*)&Vs_t[n][k2] = 0u;       // zero-pad K rows 48..63
    }
#if HAVE_ASYNC_LDS
    __builtin_amdgcn_s_wait_asynccnt(0);
#endif
    __syncthreads();

    v8f zero = {};
    v8f acc[3] = {zero, zero, zero};
    #pragma unroll
    for (int k0 = 0; k0 < 64; k0 += 32) {
        BF16x16 bfg;
        const int col = wave * 16 + l16;
        bfg.v = ld_frag2(&Vs_t[col][k0 + lh * 8], &Vs_t[col][k0 + 16 + lh * 8]);
        #pragma unroll
        for (int mt = 0; mt < 3; ++mt) {
            BF16x16 afg;
            const int row = mt * 16 + l16;
            afg.v = ld_frag2(&Asm[row][k0 + lh * 8], &Asm[row][k0 + 16 + lh * 8]);
            acc[mt] = wmma_bf16(afg.v, bfg.v, acc[mt]);
        }
    }

    u16* ob = O + ((size_t)bi * 384 + head * 48) * 16384;
    const int n = n0 + wave * 16 + l16;
    #pragma unroll
    for (int mt = 0; mt < 3; ++mt) {
        const int cb = mt * 16 + lh * 8;
        #pragma unroll
        for (int r = 0; r < 8; ++r)
            ob[(size_t)(cb + r) * 16384 + n] = f2bf_u(acc[mt][r]);
    }
}

// ---------------------------------------------------------------------------
// Launch: x -> [gemm 384->1152, bf16 out] -> [dw3x3 bf16] -> [scores]
//          -> [apply, bf16 out] -> [gemm proj, bf16 in / f32 out]
// Workspace (bf16 intermediates): qkv_pre 302MB | qkv_post 302MB.
// After dwconv, qkv_pre region is dead -> reused for attn matrices (offset 0)
// and the bf16 attention output (offset 16MB, 100MB).
// ---------------------------------------------------------------------------
extern "C" void kernel_launch(void* const* d_in, const int* in_sizes, int n_in,
                              void* d_out, int out_size, void* d_ws, size_t ws_size,
                              hipStream_t stream) {
    (void)in_sizes; (void)n_in; (void)out_size; (void)ws_size;
    const float* x           = (const float*)d_in[0];
    const float* qkv_w       = (const float*)d_in[1];
    const float* dw_w        = (const float*)d_in[2];
    const float* proj_w      = (const float*)d_in[3];
    const float* temperature = (const float*)d_in[4];

    const int B = 8, DIM = 384, QKV = 1152, HW = 16384;
    const size_t qkvElems = (size_t)B * QKV * HW;

    u16* qkv_pre  = (u16*)d_ws;                                  // 302 MB
    u16* qkv_post = qkv_pre + qkvElems;                          // 302 MB
    u16* attn_buf = (u16*)d_ws;                                  // 384 KB (reuse)
    u16* attn_o   = (u16*)((char*)d_ws + (16u << 20));           // 100 MB (reuse)

    dim3 g1(HW / 128, QKV / 128, B);
    gemm_w_x<0, 1><<<g1, 256, 0, stream>>>(qkv_w, x, qkv_pre, QKV, DIM, HW);

    const size_t total = (size_t)B * QKV * HW;
    dwconv3x3<<<(unsigned)(total / 256), 256, 0, stream>>>(qkv_pre, dw_w, qkv_post);

    attn_score<<<64, 256, 0, stream>>>(qkv_post, temperature, attn_buf);

    dim3 g3(HW / 128, 8, 8);
    attn_apply<<<g3, 256, 0, stream>>>(qkv_post, attn_buf, attn_o);

    dim3 g4(HW / 128, DIM / 128, B);
    gemm_w_x<1, 0><<<g4, 256, 0, stream>>>(proj_w, attn_o, (float*)d_out, DIM, DIM, HW);
}
